// Encoder_Block_76304388980799
// MI455X (gfx1250) — matrix operand order repaired and checked by reading, not compiled, not run on hardware
//
#include <hip/hip_runtime.h>

// ---------------------------------------------------------------------------
// Types for CDNA5 WMMA (wave32). gfx1250: v_wmma_f32_16x16x32_bf16
// ---------------------------------------------------------------------------
typedef __bf16 bf16_t;
typedef __attribute__((ext_vector_type(16))) __bf16 v16bf;
typedef __attribute__((ext_vector_type(8)))  float  v8f;
typedef __attribute__((ext_vector_type(8)))  unsigned int v8u;

// A-fragment K-pair base (16-bit A matrix 16x32 layout, ISA 7.12.2):
// VGPR v holds K = {kb, kb+1}, kb = (v<4 ? 2v : 16+2(v-4)) + 8*half
__device__ __forceinline__ int kbA(int v, int half) {
    return ((v < 4) ? (2 * v) : (16 + 2 * (v - 4))) + 8 * half;
}
// B-fragment K base (same K order as A fragment):
// kb = (v<4 ? 2v : 16+2(v-4)) + 8*half
__device__ __forceinline__ int kbB(int v, int half) {
    return ((v < 4) ? (2 * v) : (16 + 2 * (v - 4))) + 8 * half;
}

// ---------------------------------------------------------------------------
// Async global->LDS copy (16B per lane), CDNA5 async path (ASYNCcnt-tracked).
// Builtin signature (from hipcc diagnostic): (v4i as1*, v4i as3*, imm, imm).
// ---------------------------------------------------------------------------
typedef int v4i_vs __attribute__((vector_size(16)));

__device__ __forceinline__ void async_copy_b128(const void* g, void* l) {
#if defined(__gfx1250__)
#if __has_builtin(__builtin_amdgcn_global_load_async_to_lds_b128)
    typedef __attribute__((address_space(1))) v4i_vs GV4;
    typedef __attribute__((address_space(3))) v4i_vs LV4;
    // inttoptr casts: generic LDS pointer's low 32 bits == LDS byte offset
    __builtin_amdgcn_global_load_async_to_lds_b128(
        (GV4*)(unsigned long long)(uintptr_t)g,
        (LV4*)(unsigned)(uintptr_t)l, 0, 0);
#else
    unsigned lds = (unsigned)(uintptr_t)l;
    asm volatile("global_load_async_to_lds_b128 %0, %1, off"
                 :: "v"(lds), "v"(g) : "memory");
#endif
#else
    *reinterpret_cast<uint4*>(l) = *reinterpret_cast<const uint4*>(g);
#endif
}

__device__ __forceinline__ void wait_async0() {
#if defined(__gfx1250__)
#if __has_builtin(__builtin_amdgcn_s_wait_asynccnt)
    __builtin_amdgcn_s_wait_asynccnt(0);
#else
    asm volatile("s_wait_asynccnt 0" ::: "memory");
#endif
#endif
}

// ---------------------------------------------------------------------------
// Packing kernels (one-time, tiny vs 23.3 TB/s)
// ---------------------------------------------------------------------------
// f32 [K,N] row-major -> bf16 [N,K] row-major (transposed weight)
__global__ void transpose_cast(const float* __restrict__ in, bf16_t* __restrict__ out,
                               int K, int N) {
    int i = blockIdx.x * blockDim.x + threadIdx.x;   // over N*K, output-linear
    if (i >= N * K) return;
    int n = i / K, k = i - n * K;
    out[i] = (bf16_t)in[(size_t)k * N + n];
}

// W [16,1024,64] f32 -> out rows (h*64+e), cols d, ld 1024 (transposed, packed)
__global__ void pack_head_w_t(const float* __restrict__ W, bf16_t* __restrict__ out) {
    int i = blockIdx.x * blockDim.x + threadIdx.x;   // over 16*1024*64, output-linear
    if (i >= 16 * 1024 * 64) return;
    int r = i >> 10;          // h*64+e
    int d = i & 1023;
    int h = r >> 6, e = r & 63;
    out[i] = (bf16_t)W[h * 65536 + d * 64 + e];
}

__global__ void cast_f32_bf16(const float* __restrict__ in, bf16_t* __restrict__ out, int n) {
    int i = blockIdx.x * blockDim.x + threadIdx.x;
    if (i < n) out[i] = (bf16_t)in[i];
}

// QKV packed [B*S, 3072] -> Vt[b,h,e,s] (V section transposed per head)
__global__ void v_transpose(const bf16_t* __restrict__ QKV, bf16_t* __restrict__ Vt) {
    int i = blockIdx.x * blockDim.x + threadIdx.x;   // over 2*16*64*2048
    if (i >= 2 * 16 * 64 * 2048) return;
    int s = i & 2047;
    int e = (i >> 11) & 63;
    int h = (i >> 17) & 15;
    int b = (i >> 21) & 1;
    Vt[i] = QKV[(size_t)(b * 2048 + s) * 3072 + 2048 + h * 64 + e];
}

// ---------------------------------------------------------------------------
// bf16 WMMA GEMM: out[M,N] = A[M,K](rm) x Bt[N,K](rm, pre-transposed weight)
// 64x64 block tile, 128 threads (4 waves), each wave a 32x32 quadrant.
// Double-buffered async global->LDS staging; all fragment reads contiguous.
// ---------------------------------------------------------------------------
__global__ __launch_bounds__(128)
void gemm_bf16_wmma(const bf16_t* __restrict__ A, int lda,
                    const bf16_t* __restrict__ Bt, int N,
                    float* outF, bf16_t* outB, const float* __restrict__ bias,
                    int K, int relu)
{
    __shared__ __align__(16) unsigned short As[2][64][40];  // [m][k] row 80B
    __shared__ __align__(16) unsigned short Bs[2][64][40];  // [n][k] row 80B

    const int tid  = threadIdx.x;
    const int lane = tid & 31;
    const int wave = tid >> 5;
    const int l15  = lane & 15;
    const int half = lane >> 4;
    const int m0 = blockIdx.y * 64;
    const int n0 = blockIdx.x * 64;
    const int waveM = (wave & 1) * 32;
    const int waveN = (wave >> 1) * 32;

    auto stage = [&](int buf, int k0) {
#pragma unroll
        for (int j = 0; j < 2; ++j) {      // A tile: 64 rows x 32 k
            int idx = tid + j * 128;
            int m  = idx >> 2;
            int kk = (idx & 3) * 8;
            async_copy_b128(A + (size_t)(m0 + m) * lda + k0 + kk, &As[buf][m][kk]);
        }
#pragma unroll
        for (int j = 0; j < 2; ++j) {      // B tile: 64 n-rows x 32 k
            int idx = tid + j * 128;
            int n  = idx >> 2;
            int kk = (idx & 3) * 8;
            async_copy_b128(Bt + (size_t)(n0 + n) * K + k0 + kk, &Bs[buf][n][kk]);
        }
    };

    v8f acc[2][2] = {};
    stage(0, 0);
    int cur = 0;

    for (int k0 = 0; k0 < K; k0 += 32) {
        wait_async0();
        __syncthreads();
        if (k0 + 32 < K) stage(cur ^ 1, k0 + 32);

        v8u afr[2], bfr[2];
#pragma unroll
        for (int mi = 0; mi < 2; ++mi) {
            int row = waveM + mi * 16 + l15;
#pragma unroll
            for (int v = 0; v < 8; ++v)
                afr[mi][v] = *reinterpret_cast<const unsigned int*>(
                    &As[cur][row][kbA(v, half)]);
        }
#pragma unroll
        for (int ni = 0; ni < 2; ++ni) {
            int col = waveN + ni * 16 + l15;
#pragma unroll
            for (int v = 0; v < 8; ++v)
                bfr[ni][v] = *reinterpret_cast<const unsigned int*>(
                    &Bs[cur][col][kbB(v, half)]);
        }
#pragma unroll
        for (int mi = 0; mi < 2; ++mi)
#pragma unroll
            for (int ni = 0; ni < 2; ++ni)
                acc[mi][ni] = __builtin_amdgcn_wmma_f32_16x16x32_bf16(
                    false, __builtin_bit_cast(v16bf, afr[mi]),
                    false, __builtin_bit_cast(v16bf, bfr[ni]),
                    (short)0, acc[mi][ni], false, false);
        cur ^= 1;
    }

    // epilogue: C layout = lane l: N=l&15; VGPR r: M=r+8*(l>>4)
#pragma unroll
    for (int mi = 0; mi < 2; ++mi) {
#pragma unroll
        for (int ni = 0; ni < 2; ++ni) {
            int col = n0 + waveN + ni * 16 + l15;
            float bv = bias ? bias[col] : 0.0f;
#pragma unroll
            for (int r = 0; r < 8; ++r) {
                int row = m0 + waveM + mi * 16 + r + 8 * half;
                float v = acc[mi][ni][r] + bv;
                if (relu) v = fmaxf(v, 0.0f);
                size_t oidx = (size_t)row * (size_t)N + col;
                if (outF) outF[oidx] = v;
                if (outB) outB[oidx] = (bf16_t)v;
            }
        }
    }
}

// ---------------------------------------------------------------------------
// Flash attention (bf16 WMMA, fp32 online softmax).
// QKV packed [B*S, 3072] (Q cols 0..1023, K cols 1024..2047), Vt[b,h,e,s].
// Grid: (S/64, H, B); 128 threads = 4 waves; wave w handles 16 query rows.
// Key-padding mask: keys >= len_chair[b] -> -1e9; fully-masked tiles skipped.
// ---------------------------------------------------------------------------
__global__ __launch_bounds__(128)
void flash_attn(const bf16_t* __restrict__ QKV, const bf16_t* __restrict__ Vt,
                const int* __restrict__ len_chair,
                bf16_t* __restrict__ heads /* [B*S, 1024], col = h*64+e */)
{
    __shared__ __align__(16) unsigned short Ks[2][32][72];  // [key][dim]
    __shared__ __align__(16) unsigned short Vs[2][64][40];  // [dim][key]
    __shared__ __align__(16) unsigned short Ps[4][16][36];  // per-wave probs

    const int tid = threadIdx.x, lane = tid & 31, wave = tid >> 5;
    const int l15 = lane & 15, half = lane >> 4;
    const int b = blockIdx.z, h = blockIdx.y;
    const int q0 = blockIdx.x * 64 + wave * 16;
    const int len = len_chair[b];
    const size_t rs = 3 * 1024;
    const bf16_t* Qb  = QKV + (size_t)b * 2048 * rs + h * 64;
    const bf16_t* Kb  = Qb + 1024;
    const bf16_t* Vth = Vt + ((size_t)(b * 16 + h)) * 64 * 2048;   // [dim][s]

    // preload Q A-fragments (DH=64 => two K=32 chunks)
    v8u qfr[2];
#pragma unroll
    for (int c = 0; c < 2; ++c)
#pragma unroll
        for (int v = 0; v < 8; ++v)
            qfr[c][v] = *reinterpret_cast<const unsigned int*>(
                Qb + (size_t)(q0 + l15) * rs + c * 32 + kbA(v, half));

    auto stageKV = [&](int buf, int kb0) {
#pragma unroll
        for (int j = 0; j < 2; ++j) {      // K tile: 32 keys x 64 dims
            int idx = tid + j * 128;
            int key = idx >> 3;
            int dd  = (idx & 7) * 8;
            async_copy_b128(Kb + (size_t)(kb0 + key) * rs + dd, &Ks[buf][key][dd]);
        }
#pragma unroll
        for (int j = 0; j < 2; ++j) {      // V tile: 64 dims x 32 keys (from Vt)
            int idx = tid + j * 128;
            int dim = idx >> 2;
            int kk  = (idx & 3) * 8;
            async_copy_b128(Vth + (size_t)dim * 2048 + kb0 + kk, &Vs[buf][dim][kk]);
        }
    };

    float m_r[8], l_r[8];
    v8f oacc[4] = {};
#pragma unroll
    for (int r = 0; r < 8; ++r) { m_r[r] = -3.0e38f; l_r[r] = 0.0f; }

    int ntiles = (len + 31) / 32;
    if (ntiles > 64) ntiles = 64;
    stageKV(0, 0);
    int cur = 0;

    for (int t = 0; t < ntiles; ++t) {
        wait_async0();
        __syncthreads();
        if (t + 1 < ntiles) stageKV(cur ^ 1, (t + 1) * 32);
        const int kb0 = t * 32;

        // scores: 16 q-rows x 32 keys = two 16x16 tiles, K-dim=64 (2 chunks)
        v8f sc[2] = {};
#pragma unroll
        for (int nt = 0; nt < 2; ++nt) {
#pragma unroll
            for (int c = 0; c < 2; ++c) {
                v8u bfr;
#pragma unroll
                for (int v = 0; v < 8; ++v)
                    bfr[v] = *reinterpret_cast<const unsigned int*>(
                        &Ks[cur][nt * 16 + l15][c * 32 + kbB(v, half)]);
                sc[nt] = __builtin_amdgcn_wmma_f32_16x16x32_bf16(
                    false, __builtin_bit_cast(v16bf, qfr[c]),
                    false, __builtin_bit_cast(v16bf, bfr),
                    (short)0, sc[nt], false, false);
            }
        }
        // scale 1/sqrt(64) + key-padding mask
#pragma unroll
        for (int nt = 0; nt < 2; ++nt) {
            bool msk = (kb0 + nt * 16 + l15) >= len;
#pragma unroll
            for (int r = 0; r < 8; ++r)
                sc[nt][r] = msk ? -1e9f : (sc[nt][r] * 0.125f);
        }
        // online softmax; row = r + 8*half, spread over 16 lanes
#pragma unroll
        for (int r = 0; r < 8; ++r) {
            float tmax = fmaxf(sc[0][r], sc[1][r]);
#pragma unroll
            for (int off = 8; off >= 1; off >>= 1)
                tmax = fmaxf(tmax, __shfl_xor(tmax, off, 16));
            float mnew  = fmaxf(m_r[r], tmax);
            float alpha = __expf(m_r[r] - mnew);
            float p0 = __expf(sc[0][r] - mnew);
            float p1 = __expf(sc[1][r] - mnew);
            float ps = p0 + p1;
#pragma unroll
            for (int off = 8; off >= 1; off >>= 1)
                ps += __shfl_xor(ps, off, 16);
            l_r[r] = l_r[r] * alpha + ps;
            m_r[r] = mnew;
#pragma unroll
            for (int tt = 0; tt < 4; ++tt) oacc[tt][r] *= alpha;
            Ps[wave][r + 8 * half][l15]      = __builtin_bit_cast(unsigned short, (bf16_t)p0);
            Ps[wave][r + 8 * half][16 + l15] = __builtin_bit_cast(unsigned short, (bf16_t)p1);
        }
        // P (16x32) x V (32x64): A-frag from Ps, B-frags from Vs[dim][key]
        v8u pfr;
#pragma unroll
        for (int v = 0; v < 8; ++v)
            pfr[v] = *reinterpret_cast<const unsigned int*>(&Ps[wave][l15][kbA(v, half)]);
#pragma unroll
        for (int tt = 0; tt < 4; ++tt) {
            v8u vfr;
#pragma unroll
            for (int v = 0; v < 8; ++v)
                vfr[v] = *reinterpret_cast<const unsigned int*>(
                    &Vs[cur][tt * 16 + l15][kbB(v, half)]);
            oacc[tt] = __builtin_amdgcn_wmma_f32_16x16x32_bf16(
                false, __builtin_bit_cast(v16bf, pfr),
                false, __builtin_bit_cast(v16bf, vfr),
                (short)0, oacc[tt], false, false);
        }
        cur ^= 1;
    }

    // finalize: O /= l, store head-concat output
#pragma unroll
    for (int r = 0; r < 8; ++r) {
        float inv = 1.0f / l_r[r];
        int row = q0 + r + 8 * half;
#pragma unroll
        for (int t = 0; t < 4; ++t)
            heads[((size_t)b * 2048 + row) * 1024 + h * 64 + t * 16 + l15] =
                (bf16_t)(oacc[t][r] * inv);
    }
}

// ---------------------------------------------------------------------------
// y = LayerNorm(lin)*g + b + res ; write f32 (and optional bf16 copy).
// One block (256 threads) per row of 1024.
// ---------------------------------------------------------------------------
__global__ __launch_bounds__(256)
void layernorm_residual(const float* __restrict__ lin, const float* __restrict__ g,
                        const float* __restrict__ bta, const float* __restrict__ res,
                        float* __restrict__ outF, bf16_t* outB)
{
    __shared__ float sbuf[256];
    const int row = blockIdx.x;
    const int tid = threadIdx.x;
    const float* x = lin + (size_t)row * 1024;
    float v[4]; float s = 0.0f;
#pragma unroll
    for (int i = 0; i < 4; ++i) { v[i] = x[tid + i * 256]; s += v[i]; }
    sbuf[tid] = s; __syncthreads();
    for (int off = 128; off >= 1; off >>= 1) {
        if (tid < off) sbuf[tid] += sbuf[tid + off];
        __syncthreads();
    }
    float mu = sbuf[0] * (1.0f / 1024.0f);
    __syncthreads();
    float vs = 0.0f;
#pragma unroll
    for (int i = 0; i < 4; ++i) { float d = v[i] - mu; vs += d * d; }
    sbuf[tid] = vs; __syncthreads();
    for (int off = 128; off >= 1; off >>= 1) {
        if (tid < off) sbuf[tid] += sbuf[tid + off];
        __syncthreads();
    }
    float rstd = rsqrtf(sbuf[0] * (1.0f / 1024.0f) + 1e-5f);
#pragma unroll
    for (int i = 0; i < 4; ++i) {
        int c = tid + i * 256;
        float y = (v[i] - mu) * rstd * g[c] + bta[c] + res[(size_t)row * 1024 + c];
        outF[(size_t)row * 1024 + c] = y;
        if (outB) outB[(size_t)row * 1024 + c] = (bf16_t)y;
    }
}

// ---------------------------------------------------------------------------
// Orchestration
// ---------------------------------------------------------------------------
extern "C" void kernel_launch(void* const* d_in, const int* in_sizes, int n_in,
                              void* d_out, int out_size, void* d_ws, size_t ws_size,
                              hipStream_t stream)
{
    (void)in_sizes; (void)n_in; (void)out_size; (void)ws_size;
    const float* x    = (const float*)d_in[0];
    const int*   lenc = (const int*)  d_in[1];
    const float* Wq   = (const float*)d_in[2];
    const float* Wk   = (const float*)d_in[3];
    const float* Wv   = (const float*)d_in[4];
    const float* Wo   = (const float*)d_in[5];
    const float* a1w  = (const float*)d_in[6];
    const float* a1b  = (const float*)d_in[7];
    const float* g1   = (const float*)d_in[8];
    const float* b1   = (const float*)d_in[9];
    const float* f1w  = (const float*)d_in[10];
    const float* f1b  = (const float*)d_in[11];
    const float* f2w  = (const float*)d_in[12];
    const float* f2b  = (const float*)d_in[13];
    const float* a2w  = (const float*)d_in[14];
    const float* a2b  = (const float*)d_in[15];
    const float* g2   = (const float*)d_in[16];
    const float* b2   = (const float*)d_in[17];

    const int BS = 4096, Dm = 1024, FFm = 4096;

    char* ws = (char*)d_ws;
    size_t off = 0;
    auto take = [&](size_t bytes) -> char* {
        char* p = ws + off;
        off += (bytes + 255) & ~(size_t)255;
        return p;
    };
    bf16_t* Xb    = (bf16_t*)take((size_t)BS * Dm * 2);        // reused as Att_b
    bf16_t* Wqkvb = (bf16_t*)take((size_t)3 * Dm * Dm * 2);    // [3072][1024] transposed
    bf16_t* Wob   = (bf16_t*)take((size_t)Dm * Dm * 2);        // [N][K] transposed
    bf16_t* A1wb  = (bf16_t*)take((size_t)Dm * Dm * 2);
    bf16_t* F1wb  = (bf16_t*)take((size_t)FFm * Dm * 2);       // [4096][1024]
    bf16_t* F2wb  = (bf16_t*)take((size_t)Dm * FFm * 2);       // [1024][4096]
    bf16_t* A2wb  = (bf16_t*)take((size_t)Dm * Dm * 2);
    bf16_t* QKVb  = (bf16_t*)take((size_t)BS * FFm * 2);       // QKV, reused as FF hidden
    bf16_t* Vtb   = (bf16_t*)take((size_t)2 * 16 * 64 * 2048 * 2);
    bf16_t* Headsb= (bf16_t*)take((size_t)BS * Dm * 2);
    float*  lin1  = (float*) take((size_t)BS * Dm * 4);        // reused as lin2
    float*  x1f   = (float*) take((size_t)BS * Dm * 4);
    bf16_t* X1b   = (bf16_t*)take((size_t)BS * Dm * 2);
    bf16_t* FF2b  = (bf16_t*)take((size_t)BS * Dm * 2);
    bf16_t* Attb  = Xb;
    bf16_t* FFhb  = QKVb;
    float*  lin2  = lin1;

    const int T = 256;
    const int HW = 16 * 1024 * 64;
    cast_f32_bf16<<<(BS * Dm + T - 1) / T, T, 0, stream>>>(x, Xb, BS * Dm);
    pack_head_w_t<<<(HW + T - 1) / T, T, 0, stream>>>(Wq, Wqkvb);
    pack_head_w_t<<<(HW + T - 1) / T, T, 0, stream>>>(Wk, Wqkvb + (size_t)Dm * Dm);
    pack_head_w_t<<<(HW + T - 1) / T, T, 0, stream>>>(Wv, Wqkvb + (size_t)2 * Dm * Dm);
    transpose_cast<<<(Dm * Dm + T - 1) / T, T, 0, stream>>>(Wo,  Wob,  Dm, Dm);
    transpose_cast<<<(Dm * Dm + T - 1) / T, T, 0, stream>>>(a1w, A1wb, Dm, Dm);
    transpose_cast<<<(Dm * FFm + T - 1) / T, T, 0, stream>>>(f1w, F1wb, Dm, FFm);
    transpose_cast<<<(FFm * Dm + T - 1) / T, T, 0, stream>>>(f2w, F2wb, FFm, Dm);
    transpose_cast<<<(Dm * Dm + T - 1) / T, T, 0, stream>>>(a2w, A2wb, Dm, Dm);

    dim3 blk(128);
    // QKV = X x Wqkv^T   [4096 x 3072]
    gemm_bf16_wmma<<<dim3(3 * Dm / 64, BS / 64), blk, 0, stream>>>(
        Xb, Dm, Wqkvb, 3 * Dm, nullptr, QKVb, nullptr, Dm, 0);
    // V section -> Vt[b,h,e,s]
    v_transpose<<<(2 * 16 * 64 * 2048 + T - 1) / T, T, 0, stream>>>(QKVb, Vtb);
    // attention -> heads
    flash_attn<<<dim3(2048 / 64, 16, 2), blk, 0, stream>>>(QKVb, Vtb, lenc, Headsb);
    // att_out = heads x Wo
    gemm_bf16_wmma<<<dim3(Dm / 64, BS / 64), blk, 0, stream>>>(
        Headsb, Dm, Wob, Dm, nullptr, Attb, nullptr, Dm, 0);
    // lin1 = att_out x add1_w + add1_b
    gemm_bf16_wmma<<<dim3(Dm / 64, BS / 64), blk, 0, stream>>>(
        Attb, Dm, A1wb, Dm, lin1, nullptr, a1b, Dm, 0);
    // x1 = LN(lin1) + x
    layernorm_residual<<<BS, 256, 0, stream>>>(lin1, g1, b1, x, x1f, X1b);
    // ffh = relu(x1 x ff_w1 + b1)
    gemm_bf16_wmma<<<dim3(FFm / 64, BS / 64), blk, 0, stream>>>(
        X1b, Dm, F1wb, FFm, nullptr, FFhb, f1b, Dm, 1);
    // ff2 = ffh x ff_w2 + b2
    gemm_bf16_wmma<<<dim3(Dm / 64, BS / 64), blk, 0, stream>>>(
        FFhb, FFm, F2wb, Dm, nullptr, FF2b, f2b, FFm, 0);
    // lin2 = ff2 x add2_w + add2_b
    gemm_bf16_wmma<<<dim3(Dm / 64, BS / 64), blk, 0, stream>>>(
        FF2b, Dm, A2wb, Dm, lin2, nullptr, a2b, Dm, 0);
    // out = LN(lin2) + x1
    layernorm_residual<<<BS, 256, 0, stream>>>(lin2, g2, b2, x1f, (float*)d_out, nullptr);
}
